// CausalSelfAttention_6124623364279
// MI455X (gfx1250) — matrix operand order repaired and checked by reading, not compiled, not run on hardware
//
#include <hip/hip_runtime.h>

#define B_    2
#define T_    2048
#define DMOD  1024
#define NHEAD 16
#define HDIM  64
#define BNECK 1024
#define N3    3072
#define MTOT  4096   // B_*T_

typedef __attribute__((ext_vector_type(16))) __bf16 v16bf;
typedef __attribute__((ext_vector_type(8)))  float  v8f;

union Frag { v16bf v; uint4 q[2]; };

__device__ __forceinline__ unsigned short f2bf_raw(float f) {
  unsigned int u = __float_as_uint(f);
  u += 0x7fffu + ((u >> 16) & 1u);   // round-to-nearest-even
  return (unsigned short)(u >> 16);
}

__device__ __forceinline__ v8f wmma_bf16(const Frag& a, const Frag& b, v8f c) {
  return __builtin_amdgcn_wmma_f32_16x16x32_bf16(false, a.v, false, b.v,
                                                 (short)0, c, false, false);
}

// ---------------- conversion kernels ----------------
__global__ void cvt_rowmajor(const float* __restrict__ in,
                             unsigned short* __restrict__ out, int n) {
  int i = blockIdx.x * blockDim.x + threadIdx.x;
  if (i < n) out[i] = f2bf_raw(in[i]);
}

// in: [K][N] fp32 row-major  ->  out: [N][K] bf16 row-major
__global__ void cvt_transpose(const float* __restrict__ in,
                              unsigned short* __restrict__ out, int K, int N) {
  int i = blockIdx.x * blockDim.x + threadIdx.x;
  if (i < K * N) {
    int k = i % K;
    int n = i / K;
    out[(size_t)n * K + k] = f2bf_raw(in[(size_t)k * N + n]);
  }
}

// ---------------- QKV projection GEMM ----------------
// C[4096,3072] = A[4096,1024] * Bt[3072,1024]^T + bias; scatter into Q/K/Vt.
// 64x64 tile per wave: 16 WMMAs per K-step on 8 fragment loads.
__global__ __launch_bounds__(128) void qkv_gemm(
    const unsigned short* __restrict__ Ab,    // x bf16 [4096][1024]
    const unsigned short* __restrict__ Bt,    // WqkvT bf16 [3072][1024]
    const float* __restrict__ bias,           // [3072]
    unsigned short* __restrict__ Qo,          // [B*H][T][64]  (pre-scaled)
    unsigned short* __restrict__ Ko,          // [B*H][T][64]
    unsigned short* __restrict__ Vt)          // [B*H][64][T]
{
  const int lane = threadIdx.x & 31;
  const int wave = threadIdx.x >> 5;
  const int mloc = lane & 15;
  const int hi   = lane >> 4;
  const int m0 = blockIdx.x * 64;
  const int n0 = (blockIdx.y * 4 + wave) * 64;

  v8f acc[4][4];
#pragma unroll
  for (int i = 0; i < 4; i++)
#pragma unroll
    for (int j = 0; j < 4; j++) acc[i][j] = {};

  for (int k0 = 0; k0 < DMOD; k0 += 32) {
    Frag a[4], b[4];
#pragma unroll
    for (int i = 0; i < 4; i++) {
      const uint4* p = (const uint4*)(Ab + (size_t)(m0 + i * 16 + mloc) * DMOD + k0 + hi * 8);
      a[i].q[0] = p[0]; a[i].q[1] = p[2];
    }
#pragma unroll
    for (int j = 0; j < 4; j++) {
      const uint4* p = (const uint4*)(Bt + (size_t)(n0 + j * 16 + mloc) * DMOD + k0 + hi * 8);
      b[j].q[0] = p[0]; b[j].q[1] = p[2];
    }
#pragma unroll
    for (int i = 0; i < 4; i++)
#pragma unroll
      for (int j = 0; j < 4; j++)
        acc[i][j] = wmma_bf16(a[i], b[j], acc[i][j]);
  }

#pragma unroll
  for (int im = 0; im < 4; im++) {
#pragma unroll
    for (int jn = 0; jn < 4; jn++) {
      int ncol = n0 + jn * 16 + mloc;
      float bv = bias[ncol];
      int which = ncol >> 10;            // 0=q 1=k 2=v
      int h = (ncol >> 6) & 15;
      int d = ncol & 63;
#pragma unroll
      for (int r = 0; r < 8; r++) {
        int row = m0 + im * 16 + r + hi * 8;   // global (b,t) row
        int b  = row >> 11;
        int tt = row & (T_ - 1);
        float val = acc[im][jn][r] + bv;
        size_t bh = (size_t)(b * NHEAD + h);
        if (which == 0) {
          Qo[(bh * T_ + tt) * HDIM + d] = f2bf_raw(val * 0.125f); // 1/sqrt(64)
        } else if (which == 1) {
          Ko[(bh * T_ + tt) * HDIM + d] = f2bf_raw(val);
        } else {
          Vt[(bh * HDIM + d) * T_ + tt] = f2bf_raw(val);
        }
      }
    }
  }
}

// ---------------- flash attention (causal), 64-key blocks ----------------
__global__ __launch_bounds__(128) void attn(
    const unsigned short* __restrict__ Qm,  // [B*H][T][64] (pre-scaled)
    const unsigned short* __restrict__ Km,  // [B*H][T][64]
    const unsigned short* __restrict__ Vt,  // [B*H][64][T]
    unsigned short* __restrict__ Y)         // [B][T][1024] bf16
{
  __shared__ unsigned short lds[4][16 * 64];
  const int lane = threadIdx.x & 31;
  const int wave = threadIdx.x >> 5;
  const int mloc = lane & 15;
  const int hi   = lane >> 4;

  int tileId = blockIdx.x * 4 + wave;       // B*H*(T/16) = 4096 tiles
  int qTile = tileId & 127;                 // T/16 = 128
  int bh    = tileId >> 7;                  // 0..31
  int h = bh & 15, b = bh >> 4;
  const unsigned short* Qp = Qm + (size_t)bh * T_ * HDIM;
  const unsigned short* Kp = Km + (size_t)bh * T_ * HDIM;
  const unsigned short* Vp = Vt + (size_t)bh * HDIM * T_;
  const int qBase = qTile * 16;

  // load Q fragments (A layout), d-chunks of 32
  Frag aQ[2];
#pragma unroll
  for (int c = 0; c < 2; c++) {
    const uint4* p = (const uint4*)(Qp + (size_t)(qBase + mloc) * HDIM + c * 32 + hi * 8);
    aQ[c].q[0] = p[0]; aQ[c].q[1] = p[2];
  }

  v8f O[4];
#pragma unroll
  for (int t = 0; t < 4; t++) O[t] = {};
  float mrow[8], lrow[8];
#pragma unroll
  for (int r = 0; r < 8; r++) { mrow[r] = -1e30f; lrow[r] = 0.f; }

  unsigned short* P = lds[wave];
  const int kEnd = qBase + 16;              // causal: keys < qBase+16

  for (int kb = 0; kb < kEnd; kb += 64) {
    v8f S[4];
#pragma unroll
    for (int j = 0; j < 4; j++) S[j] = {};

#pragma unroll
    for (int c = 0; c < 2; c++) {
#pragma unroll
      for (int j = 0; j < 4; j++) {
        Frag bK;
        const uint4* p = (const uint4*)(Kp + (size_t)(kb + j * 16 + mloc) * HDIM + c * 32 + hi * 8);
        bK.q[0] = p[0]; bK.q[1] = p[2];
        S[j] = wmma_bf16(aQ[c], bK, S[j]);
      }
    }

    // causal mask (only diagonal-straddling blocks need it; kb <= qBase always)
    if (kb + 63 > qBase) {
#pragma unroll
      for (int j = 0; j < 4; j++)
#pragma unroll
        for (int r = 0; r < 8; r++) {
          int row = qBase + r + hi * 8;
          if (kb + j * 16 + mloc > row) S[j][r] = -1e30f;
        }
    }

    // online softmax; rows live across 16-lane halves
    float corr[8];
#pragma unroll
    for (int r = 0; r < 8; r++) {
      float mx = fmaxf(fmaxf(S[0][r], S[1][r]), fmaxf(S[2][r], S[3][r]));
      for (int off = 1; off < 16; off <<= 1) mx = fmaxf(mx, __shfl_xor(mx, off, 32));
      float mnew = fmaxf(mrow[r], mx);
      corr[r] = __expf(mrow[r] - mnew);
      mrow[r] = mnew;
      float p0 = __expf(S[0][r] - mnew);
      float p1 = __expf(S[1][r] - mnew);
      float p2 = __expf(S[2][r] - mnew);
      float p3 = __expf(S[3][r] - mnew);
      float sum = (p0 + p1) + (p2 + p3);
      for (int off = 1; off < 16; off <<= 1) sum += __shfl_xor(sum, off, 32);
      lrow[r] = lrow[r] * corr[r] + sum;
      int rowl = r + hi * 8;
      P[rowl * 64 + mloc]      = f2bf_raw(p0);
      P[rowl * 64 + 16 + mloc] = f2bf_raw(p1);
      P[rowl * 64 + 32 + mloc] = f2bf_raw(p2);
      P[rowl * 64 + 48 + mloc] = f2bf_raw(p3);
    }
#pragma unroll
    for (int t = 0; t < 4; t++)
#pragma unroll
      for (int r = 0; r < 8; r++) O[t][r] *= corr[r];

    // C-layout -> A-fragment re-swizzle through LDS (same-wave DS ops are in-order)
    __builtin_amdgcn_wave_barrier();
    Frag aP[2];
#pragma unroll
    for (int c = 0; c < 2; c++) {
      const uint4* p = (const uint4*)(P + mloc * 64 + c * 32 + hi * 8);
      aP[c].q[0] = p[0]; aP[c].q[1] = p[2];
    }
    __builtin_amdgcn_wave_barrier();

#pragma unroll
    for (int c = 0; c < 2; c++) {
#pragma unroll
      for (int t = 0; t < 4; t++) {
        Frag bV;
        const uint4* p = (const uint4*)(Vp + (size_t)(t * 16 + mloc) * T_ + kb + c * 32 + hi * 8);
        bV.q[0] = p[0]; bV.q[1] = p[2];
        O[t] = wmma_bf16(aP[c], bV, O[t]);
      }
    }
  }

#pragma unroll
  for (int t = 0; t < 4; t++) {
#pragma unroll
    for (int r = 0; r < 8; r++) {
      int row = qBase + r + hi * 8;
      float val = O[t][r] / lrow[r];
      Y[((size_t)b * T_ + row) * BNECK + h * HDIM + t * 16 + mloc] = f2bf_raw(val);
    }
  }
}

// ---------------- output projection GEMM (64x64 tile per wave) ----------------
__global__ __launch_bounds__(128) void proj_gemm(
    const unsigned short* __restrict__ Ab,   // Y bf16 [4096][1024]
    const unsigned short* __restrict__ Bt,   // WprojT bf16 [1024][1024]
    const float* __restrict__ bias,          // [1024]
    float* __restrict__ out)                 // [4096][1024] fp32
{
  const int lane = threadIdx.x & 31;
  const int wave = threadIdx.x >> 5;
  const int mloc = lane & 15;
  const int hi   = lane >> 4;
  const int m0 = blockIdx.x * 64;
  const int n0 = (blockIdx.y * 4 + wave) * 64;

  v8f acc[4][4];
#pragma unroll
  for (int i = 0; i < 4; i++)
#pragma unroll
    for (int j = 0; j < 4; j++) acc[i][j] = {};

  for (int k0 = 0; k0 < BNECK; k0 += 32) {
    Frag a[4], b[4];
#pragma unroll
    for (int i = 0; i < 4; i++) {
      const uint4* p = (const uint4*)(Ab + (size_t)(m0 + i * 16 + mloc) * BNECK + k0 + hi * 8);
      a[i].q[0] = p[0]; a[i].q[1] = p[2];
    }
#pragma unroll
    for (int j = 0; j < 4; j++) {
      const uint4* p = (const uint4*)(Bt + (size_t)(n0 + j * 16 + mloc) * BNECK + k0 + hi * 8);
      b[j].q[0] = p[0]; b[j].q[1] = p[2];
    }
#pragma unroll
    for (int i = 0; i < 4; i++)
#pragma unroll
      for (int j = 0; j < 4; j++)
        acc[i][j] = wmma_bf16(a[i], b[j], acc[i][j]);
  }

#pragma unroll
  for (int im = 0; im < 4; im++) {
#pragma unroll
    for (int jn = 0; jn < 4; jn++) {
      int ncol = n0 + jn * 16 + mloc;
      float bv = bias[ncol];
#pragma unroll
      for (int r = 0; r < 8; r++) {
        int row = m0 + im * 16 + r + hi * 8;
        out[(size_t)row * DMOD + ncol] = acc[im][jn][r] + bv;
      }
    }
  }
}

// ---------------- host launcher ----------------
extern "C" void kernel_launch(void* const* d_in, const int* in_sizes, int n_in,
                              void* d_out, int out_size, void* d_ws, size_t ws_size,
                              hipStream_t stream) {
  (void)in_sizes; (void)n_in; (void)out_size; (void)ws_size;
  const float* x     = (const float*)d_in[0];
  const float* Wqkv  = (const float*)d_in[1];
  const float* bqkv  = (const float*)d_in[2];
  const float* Wproj = (const float*)d_in[3];
  const float* bproj = (const float*)d_in[4];
  float* out = (float*)d_out;

  char* ws = (char*)d_ws;
  size_t off = 0;
  auto take = [&](size_t bytes) { size_t o = off; off += (bytes + 255) & ~(size_t)255; return o; };

  unsigned short* xb     = (unsigned short*)(ws + take((size_t)MTOT * DMOD * 2));
  unsigned short* wqkvT  = (unsigned short*)(ws + take((size_t)N3 * DMOD * 2));
  unsigned short* wprojT = (unsigned short*)(ws + take((size_t)BNECK * DMOD * 2));
  unsigned short* Qb     = (unsigned short*)(ws + take((size_t)B_ * NHEAD * T_ * HDIM * 2));
  unsigned short* Kb     = (unsigned short*)(ws + take((size_t)B_ * NHEAD * T_ * HDIM * 2));
  unsigned short* Vtb    = (unsigned short*)(ws + take((size_t)B_ * NHEAD * HDIM * T_ * 2));
  unsigned short* Yb     = (unsigned short*)(ws + take((size_t)MTOT * BNECK * 2));

  int nx = MTOT * DMOD;
  cvt_rowmajor<<<(nx + 255) / 256, 256, 0, stream>>>(x, xb, nx);
  cvt_transpose<<<((DMOD * N3) + 255) / 256, 256, 0, stream>>>(Wqkv, wqkvT, DMOD, N3);
  cvt_transpose<<<((DMOD * BNECK) + 255) / 256, 256, 0, stream>>>(Wproj, wprojT, BNECK, BNECK);

  qkv_gemm<<<dim3(MTOT / 64, N3 / 256), 128, 0, stream>>>(xb, wqkvT, bqkv, Qb, Kb, Vtb);
  attn<<<dim3((B_ * NHEAD * (T_ / 16)) / 4), 128, 0, stream>>>(Qb, Kb, Vtb, Yb);
  proj_gemm<<<dim3(MTOT / 64, BNECK / 256), 128, 0, stream>>>(Yb, wprojT, bproj, out);
}
